// ESN_mlr_5394478924038
// MI455X (gfx1250) — matrix operand order repaired and checked by reading, not compiled, not run on hardware
//
#include <hip/hip_runtime.h>
#include <hip/hip_bf16.h>

// ---------------- types ----------------
typedef __bf16 bf16_t;
typedef __attribute__((ext_vector_type(16))) __bf16 v16bf;
typedef __attribute__((ext_vector_type(8)))  __bf16 v8bf;
typedef __attribute__((ext_vector_type(8)))  float  v8f;
typedef __attribute__((ext_vector_type(4))) unsigned int u32x4;
typedef __attribute__((ext_vector_type(8))) int i32x8;
typedef __attribute__((ext_vector_type(4))) int i32x4;

// ---------------- problem constants ----------------
constexpr int kVocab = 32000;
constexpr int kN     = 4096;    // reservoir size
constexpr int kROut  = 512;
constexpr int kBatch = 8;
constexpr int kT     = 256;
constexpr int kNNZ   = kN * 32; // 131072

// =====================================================================
// Phase 0: f32 -> bf16 weight conversion
// =====================================================================
__global__ void cvt_f32_bf16(const float* __restrict__ s, bf16_t* __restrict__ d, int n) {
    int i = blockIdx.x * blockDim.x + threadIdx.x;
    if (i < n) d[i] = (bf16_t)s[i];
}

// =====================================================================
// Phase 1: sequential ESN recurrence, one workgroup per batch element.
// State h lives in LDS; COO sparse matvec uses ds_add_f32 LDS atomics.
// Emits H in bf16, row index r = b*T + t.
// =====================================================================
__global__ __launch_bounds__(1024) void esn_recurrence(
    const int*   __restrict__ x,       // (B, T)
    const float* __restrict__ W_in,    // (VOCAB, N)
    const int*   __restrict__ rrows,   // (NNZ)
    const int*   __restrict__ rcols,   // (NNZ)
    const float* __restrict__ rvals,   // (NNZ)
    const float* __restrict__ leak,    // (N)
    bf16_t*      __restrict__ H_bf)    // (B*T, N) bf16
{
    __shared__ float h_s[kN];
    __shared__ float rec_s[kN];
    __shared__ float a_s[kN];

    const int b   = blockIdx.x;
    const int tid = threadIdx.x;

    for (int n = tid; n < kN; n += 1024) { h_s[n] = 0.0f; a_s[n] = leak[n]; }
    __syncthreads();

    for (int t = 0; t < kT; ++t) {
        // rec <- u  (input projection == row gather of W_in)
        const int xt = x[b * kT + t];
        const float* wrow = W_in + (long)xt * kN;
        for (int n = tid; n < kN; n += 1024) rec_s[n] = wrow[n];
        __syncthreads();

        // rec += W_rec @ h   (COO scatter-add via LDS float atomics)
        for (int i = tid; i < kNNZ; i += 1024) {
            const float contrib = rvals[i] * h_s[rcols[i]];
            atomicAdd(&rec_s[rrows[i]], contrib);
        }
        __syncthreads();

        // leaky tanh update + emit bf16 row
        bf16_t* hrow = H_bf + ((long)b * kT + t) * kN;
        for (int n = tid; n < kN; n += 1024) {
            float pre = rec_s[n];
            pre = fminf(fmaxf(pre, -10.0f), 10.0f);
            const float an = a_s[n];
            const float hn = (1.0f - an) * h_s[n] + an * tanhf(pre);
            h_s[n] = hn;
            hrow[n] = (bf16_t)hn;
        }
        __syncthreads();
    }
}

// =====================================================================
// Tensor Data Mover: 2D tile (bf16) Global -> LDS.
// D# per CDNA5 ISA §8: group0 {count=1 | lds_addr | global_addr | type=2},
// group1 {data_size=2B, tensor dims, 16xK tile, dim0 stride}.
// This toolchain exposes the 6-arg builtin:
//   (u32x4 g0, i32x8 g1, i32x4 g2, i32x4 g3, i32x8 extra, i32 cpol)
// =====================================================================
__device__ __forceinline__ unsigned lds_off_of(const void* p) {
    // generic LDS address = {SHARED_BASE, offset}; low 32 bits are the LDS offset
    return (unsigned)(unsigned long long)p;
}

__device__ __forceinline__ void tdm_load_2d_bf16(
    unsigned lds_off, const void* gaddr,
    unsigned tensor_d0, unsigned tensor_d1,
    unsigned tile_d0, unsigned tile_d1, unsigned stride0)
{
    const unsigned long long ga = (unsigned long long)gaddr;
    u32x4 g0;
    g0[0] = 1u;                                              // count=1, user D#
    g0[1] = lds_off;                                         // lds_addr (bytes)
    g0[2] = (unsigned)ga;                                    // global_addr[31:0]
    g0[3] = (unsigned)((ga >> 32) & 0x01ffffffu) | (2u << 30); // addr[56:32] | type=2
    i32x8 g1;
    g1[0] = 0x00010000;                                      // data_size=1 (2 bytes)
    g1[1] = (int)(tensor_d0 << 16);                          // tensor_dim0[15:0]
    g1[2] = (int)((tensor_d0 >> 16) | (tensor_d1 << 16));    // dim0 hi | dim1 lo
    g1[3] = (int)((tensor_d1 >> 16) | (tile_d0 << 16));      // dim1 hi | tile_dim0
    g1[4] = (int)(tile_d1 & 0xffffu);                        // tile_dim1 (tile_dim2=0)
    g1[5] = (int)stride0;                                    // tensor_dim0_stride lo32
    g1[6] = 0;
    g1[7] = 0;
    const i32x4 z4 = {0, 0, 0, 0};
    const i32x8 z8 = {0, 0, 0, 0, 0, 0, 0, 0};
    __builtin_amdgcn_tensor_load_to_lds(g0, g1, z4, z4, z8, 0);
}

// =====================================================================
// WMMA fragment helpers (v_wmma_f32_16x16x32_bf16 layouts, wave32)
// =====================================================================
// B: 32x16 (KxN) where B[k][n] = Bsrc[n][k] (Bsrc row-major, leading dim ldb).
__device__ __forceinline__ v16bf asm_frag(v8bf lo, v8bf hi) {
    v16bf r;
#pragma unroll
    for (int i = 0; i < 8; ++i) { r[i] = lo[i]; r[i + 8] = hi[i]; }
    return r;
}

__device__ __forceinline__ v16bf load_b_frag(const bf16_t* __restrict__ Bsrc, int ldb,
                                             int nbase, int kbase, int lane) {
    const int col   = nbase + (lane & 15);
    const int khalf = (lane >> 4) * 8;
    const bf16_t* bp = Bsrc + (long)col * ldb + kbase + khalf;
    return asm_frag(*(const v8bf*)bp, *(const v8bf*)(bp + 16));
}

__device__ __forceinline__ v8f wmma_bf16(v16bf a, v16bf b, v8f c) {
    return __builtin_amdgcn_wmma_f32_16x16x32_bf16(
        false, a, false, b, (short)0, c, false, false);
}

// =====================================================================
// Phase 2: Z(2048x512) = H(2048x4096) @ B_w^T.
// Block = one 16-row mtile; 8 waves x 64 cols. A-tile staged in LDS by TDM,
// K chunked 4x1024 with LDS double buffering overlapped with compute.
// =====================================================================
constexpr int kKCH = 1024;

__global__ __launch_bounds__(256) void gemm_hb(
    const bf16_t* __restrict__ H,   // (2048, 4096)
    const bf16_t* __restrict__ Bw,  // (512, 4096)
    bf16_t*       __restrict__ Z)   // (2048, 512)
{
    __shared__ bf16_t As[2][16 * kKCH];  // 2 x 32 KiB

    const int lane  = threadIdx.x & 31;
    const int wave  = threadIdx.x >> 5;
    const int mbase = blockIdx.x * 16;
    const int nbase = wave * 64;

    if (threadIdx.x < 32) {
        tdm_load_2d_bf16(lds_off_of(&As[0][0]), H + (long)mbase * kN,
                         kN, 2048, kKCH, 16, kN);
        __builtin_amdgcn_s_wait_tensorcnt(0);
    }
    __syncthreads();

    const v8f zero = {};
    v8f c0 = zero, c1 = zero, c2 = zero, c3 = zero;
    const int half = lane >> 4;
    const int arow = lane & 15;

    for (int ch = 0; ch < kN / kKCH; ++ch) {
        if (ch + 1 < kN / kKCH && threadIdx.x < 32) {
            tdm_load_2d_bf16(lds_off_of(&As[(ch + 1) & 1][0]),
                             H + (long)mbase * kN + (ch + 1) * kKCH,
                             kN, 2048, kKCH, 16, kN);
        }
        const int kg0 = ch * kKCH;
        v16bf bc0 = load_b_frag(Bw, kN, nbase +  0, kg0, lane);
        v16bf bc1 = load_b_frag(Bw, kN, nbase + 16, kg0, lane);
        v16bf bc2 = load_b_frag(Bw, kN, nbase + 32, kg0, lane);
        v16bf bc3 = load_b_frag(Bw, kN, nbase + 48, kg0, lane);
#pragma unroll 4
        for (int kk = 0; kk < kKCH; kk += 32) {
            // A fragment from LDS (ds_load), stride kKCH within the staged tile
            const bf16_t* ap = &As[ch & 1][arow * kKCH + kk + half * 8];
            const v16bf a = asm_frag(*(const v8bf*)ap, *(const v8bf*)(ap + 16));
            v16bf bn0 = bc0, bn1 = bc1, bn2 = bc2, bn3 = bc3;
            if (kk + 32 < kKCH) {
                bn0 = load_b_frag(Bw, kN, nbase +  0, kg0 + kk + 32, lane);
                bn1 = load_b_frag(Bw, kN, nbase + 16, kg0 + kk + 32, lane);
                bn2 = load_b_frag(Bw, kN, nbase + 32, kg0 + kk + 32, lane);
                bn3 = load_b_frag(Bw, kN, nbase + 48, kg0 + kk + 32, lane);
            }
            c0 = wmma_bf16(a, bc0, c0);
            c1 = wmma_bf16(a, bc1, c1);
            c2 = wmma_bf16(a, bc2, c2);
            c3 = wmma_bf16(a, bc3, c3);
            bc0 = bn0; bc1 = bn1; bc2 = bn2; bc3 = bn3;
        }
        if (threadIdx.x < 32) __builtin_amdgcn_s_wait_tensorcnt(0);
        __syncthreads();
    }

    const int mrow = mbase + half * 8;
    const int ncol = lane & 15;
    v8f acc[4] = {c0, c1, c2, c3};
#pragma unroll
    for (int j = 0; j < 4; ++j) {
        bf16_t* zp = Z + (long)mrow * kROut + nbase + j * 16 + ncol;
#pragma unroll
        for (int i = 0; i < 8; ++i) zp[(long)i * kROut] = (bf16_t)acc[j][i];
    }
}

// =====================================================================
// Phase 3: Out(2048x32000) = Z(2048x512) @ A_w^T + A_b.
// Block = one mtile x 640 cols; wave = 16x80 (5 N-tiles). Z A-tile (16x512,
// 16 KiB) staged once in LDS by TDM; B streamed from L2, software-pipelined.
// Grid (50, 128) covers 32000 x 2048 exactly.
// =====================================================================
__global__ __launch_bounds__(256) void gemm_out(
    const bf16_t* __restrict__ Zm,   // (2048, 512)
    const bf16_t* __restrict__ Aw,   // (32000, 512)
    const float*  __restrict__ Ab,   // (32000)
    float*        __restrict__ out)  // (2048, 32000) == (B,T,V)
{
    __shared__ bf16_t As[16 * kROut];  // 16 KiB

    const int lane  = threadIdx.x & 31;
    const int wave  = threadIdx.x >> 5;
    const int mbase = blockIdx.y * 16;
    const int nbase = blockIdx.x * 640 + wave * 80;

    if (threadIdx.x < 32) {
        tdm_load_2d_bf16(lds_off_of(&As[0]), Zm + (long)mbase * kROut,
                         kROut, 2048, kROut, 16, kROut);
        __builtin_amdgcn_s_wait_tensorcnt(0);
    }
    __syncthreads();

    const v8f zero = {};
    v8f c[5] = {zero, zero, zero, zero, zero};
    const int half = lane >> 4;
    const int arow = lane & 15;

    v16bf bc[5];
#pragma unroll
    for (int j = 0; j < 5; ++j) bc[j] = load_b_frag(Aw, kROut, nbase + j * 16, 0, lane);

#pragma unroll 4
    for (int ks = 0; ks < kROut / 32; ++ks) {
        const int k = ks * 32;
        const bf16_t* ap = &As[arow * kROut + k + half * 8];
        const v16bf a = asm_frag(*(const v8bf*)ap, *(const v8bf*)(ap + 16));
        v16bf bn[5];
#pragma unroll
        for (int j = 0; j < 5; ++j) bn[j] = bc[j];
        if (k + 32 < kROut) {
#pragma unroll
            for (int j = 0; j < 5; ++j)
                bn[j] = load_b_frag(Aw, kROut, nbase + j * 16, k + 32, lane);
        }
#pragma unroll
        for (int j = 0; j < 5; ++j) c[j] = wmma_bf16(a, bc[j], c[j]);
#pragma unroll
        for (int j = 0; j < 5; ++j) bc[j] = bn[j];
    }

    const int mrow = mbase + half * 8;
#pragma unroll
    for (int j = 0; j < 5; ++j) {
        const int ncol = nbase + j * 16 + (lane & 15);
        const float bias = Ab[ncol];
        float* op = out + (long)mrow * kVocab + ncol;
#pragma unroll
        for (int i = 0; i < 8; ++i) op[(long)i * kVocab] = c[j][i] + bias;
    }
}

// =====================================================================
// Launch
// =====================================================================
extern "C" void kernel_launch(void* const* d_in, const int* in_sizes, int n_in,
                              void* d_out, int out_size, void* d_ws, size_t ws_size,
                              hipStream_t stream) {
    const int*   x     = (const int*)  d_in[0];
    const float* W_in  = (const float*)d_in[1];
    const int*   rrows = (const int*)  d_in[2];
    const int*   rcols = (const int*)  d_in[3];
    const float* rvals = (const float*)d_in[4];
    const float* leak  = (const float*)d_in[5];
    const float* Bw    = (const float*)d_in[6];
    const float* Aw    = (const float*)d_in[7];
    const float* Ab    = (const float*)d_in[8];
    float* out = (float*)d_out;

    // workspace layout (bytes)
    char* ws = (char*)d_ws;
    bf16_t* H_bf  = (bf16_t*)(ws);                                        // 16 MiB
    bf16_t* Z_bf  = (bf16_t*)(ws + (size_t)16777216);                     //  2 MiB
    bf16_t* Bw_bf = (bf16_t*)(ws + (size_t)16777216 + 2097152);           //  4 MiB
    bf16_t* Aw_bf = (bf16_t*)(ws + (size_t)16777216 + 2097152 + 4194304); // 31.3 MiB

    // Phase 0: weight conversion to bf16
    const int nBw = kROut * kN;      // 2,097,152
    const int nAw = kVocab * kROut;  // 16,384,000
    cvt_f32_bf16<<<(nBw + 255) / 256, 256, 0, stream>>>(Bw, Bw_bf, nBw);
    cvt_f32_bf16<<<(nAw + 255) / 256, 256, 0, stream>>>(Aw, Aw_bf, nAw);

    // Phase 1: sequential recurrence (1 WG per batch element, LDS state)
    esn_recurrence<<<kBatch, 1024, 0, stream>>>(x, W_in, rrows, rcols, rvals, leak, H_bf);

    // Phase 2: Z = H @ Bw^T   (TDM-staged A, double-buffered K chunks)
    gemm_hb<<<128, 256, 0, stream>>>(H_bf, Bw_bf, Z_bf);

    // Phase 3: Out = Z @ Aw^T + Ab   (TDM-staged A, 50x128 blocks)
    gemm_out<<<dim3(50, 128), 256, 0, stream>>>(Z_bf, Aw_bf, Ab, out);
}
